// DenseNet121_9852654977035
// MI455X (gfx1250) — 8-bit matrix operand order repaired and checked by reading, compiled, not run on hardware
//
#include <hip/hip_runtime.h>

typedef __attribute__((ext_vector_type(8))) int v8i;

__device__ __forceinline__ signed char sgn8(float v){ return (signed char)((v > 0.f) - (v < 0.f)); }

__global__ void bn_fold_k(const float* __restrict__ g, const float* __restrict__ b,
                          const float* __restrict__ m, const float* __restrict__ v,
                          float* __restrict__ sc, float* __restrict__ sh, int C) {
    int i = blockIdx.x * blockDim.x + threadIdx.x;
    if (i < C) {
        float s = g[i] * rsqrtf(v[i] + 1e-5f);
        sc[i] = s;
        sh[i] = b[i] - m[i] * s;
    }
}

__global__ void conv0_k(const float* __restrict__ x, const float* __restrict__ w,
                        const float* __restrict__ sc, const float* __restrict__ sh,
                        float* __restrict__ out) {
    int i = blockIdx.x * blockDim.x + threadIdx.x;
    const int total = 16 * 64 * 114 * 114;
    if (i >= total) return;
    int px = i % 114; int t = i / 114;
    int py = t % 114; t /= 114;
    int c = t & 63;   int n = t >> 6;
    int cnt = 0;
    for (int ky = 0; ky < 3; ky++) {
        int iy = py * 2 - 3 + ky;
        if ((unsigned)iy >= 224u) continue;
        for (int kx = 0; kx < 3; kx++) {
            int ix = px * 2 - 3 + kx;
            if ((unsigned)ix >= 224u) continue;
            float xv = x[((size_t)(n * 3 + 1) * 224 + iy) * 224 + ix];
            float wv = w[c * 9 + ky * 3 + kx];
            cnt += ((xv > 0.f) - (xv < 0.f)) * ((wv > 0.f) - (wv < 0.f));
        }
    }
    float val = sc[c] * (float)cnt + sh[c];
    out[i] = fminf(1.f, fmaxf(-1.f, val));
}

__global__ void maxpool_k(const float* __restrict__ in, float* __restrict__ out, int Cstride) {
    int i = blockIdx.x * blockDim.x + threadIdx.x;
    const int total = 16 * 64 * 57 * 57;
    if (i >= total) return;
    int xo = i % 57; int t = i / 57;
    int yo = t % 57; t /= 57;
    int c = t & 63;  int n = t >> 6;
    float v = -3.4e38f;
    for (int ky = 0; ky < 3; ky++) {
        int iy = 2 * yo - 1 + ky;
        if ((unsigned)iy >= 114u) continue;
        for (int kx = 0; kx < 3; kx++) {
            int ix = 2 * xo - 1 + kx;
            if ((unsigned)ix >= 114u) continue;
            v = fmaxf(v, in[((size_t)(n * 64 + c) * 114 + iy) * 114 + ix]);
        }
    }
    out[(((size_t)n * Cstride + c) * 57 + yo) * 57 + xo] = v;
}

__global__ void binarize_h_k(const float* __restrict__ h, const float* __restrict__ sc,
                             const float* __restrict__ sh, signed char* __restrict__ out,
                             int cin, int Kpad, int Cstride, int H, int W) {
    int i = blockIdx.x * blockDim.x + threadIdx.x;
    int HW = H * W;
    size_t total = (size_t)16 * HW * Kpad;
    if ((size_t)i >= total) return;
    int c = i % Kpad; int m = i / Kpad;
    if (c >= cin) { out[i] = 0; return; }
    int n = m / HW, r = m % HW, y = r / W, x = r % W;
    float v = h[(((size_t)n * Cstride + c) * H + y) * W + x];
    out[i] = sgn8(sc[c] * v + sh[c]);
}

__global__ void binw1_k(const float* __restrict__ w, signed char* __restrict__ out,
                        int coutPad, int coutReal, int cin, int Kpad) {
    int i = blockIdx.x * blockDim.x + threadIdx.x;
    size_t total = (size_t)coutPad * Kpad;
    if ((size_t)i >= total) return;
    int c = i % Kpad; int o = i / Kpad;
    out[i] = (o < coutReal && c < cin) ? sgn8(w[(size_t)o * cin + c]) : (signed char)0;
}

__global__ void binw2_k(const float* __restrict__ w, signed char* __restrict__ out) {
    int i = blockIdx.x * blockDim.x + threadIdx.x;
    const int total = 32 * 1152;
    if (i >= total) return;
    int k = i % 1152; int o = i / 1152;
    int tap = k / 128, c = k % 128;
    int ky = tap / 3, kx = tap % 3;
    out[i] = sgn8(w[(((size_t)o * 128 + c) * 3 + ky) * 3 + kx]);
}

template <int MODE, int NT>
__global__ void bgemm_lds_k(const signed char* __restrict__ A, const signed char* __restrict__ Bw,
                            int Mtiles, int Kpad,
                            const float* __restrict__ sc, const float* __restrict__ sh,
                            signed char* __restrict__ outB,
                            float* __restrict__ outF, int Cstride, int c0, int H, int W) {
    __shared__ int4 smem[NT * 64];
    int wave = threadIdx.x >> 5;
    int lane = threadIdx.x & 31;
    int tM = blockIdx.x * 8 + wave;
    bool active = (tM < Mtiles);
    int tileM = (active ? tM : (Mtiles - 1)) * 16;
    int tileNbase = blockIdx.y * (NT * 16);

    const signed char* arow = A + (size_t)(tileM + (lane & 15)) * Kpad + ((lane & 16) ? 16 : 0);
    int hl = (lane >> 4) & 1;
    int colin = lane & 15;

    v8i acc[NT];
#pragma unroll
    for (int s = 0; s < NT; s++) acc[s] = (v8i){0, 0, 0, 0, 0, 0, 0, 0};

    for (int k0 = 0; k0 < Kpad; k0 += 64) {
        for (int u = threadIdx.x; u < NT * 64; u += 256) {
            int col = u >> 2, kk = (u & 3) << 4;
            smem[u] = *(const int4*)(Bw + (size_t)(tileNbase + col) * Kpad + k0 + kk);
        }
        __builtin_prefetch(arow + k0 + 64, 0, 0);
        v8i a;
        const int* ap;
        ap = (const int*)(arow + k0);      a[0] = ap[0]; a[1] = ap[1];
        ap = (const int*)(arow + k0 + 8);  a[2] = ap[0]; a[3] = ap[1];
        ap = (const int*)(arow + k0 + 32); a[4] = ap[0]; a[5] = ap[1];
        ap = (const int*)(arow + k0 + 40); a[6] = ap[0]; a[7] = ap[1];
        __syncthreads();
#pragma unroll
        for (int s = 0; s < NT; s++) {
            int cb = (s * 16 + colin) * 4 + hl;
            int4 blo = smem[cb];
            int4 bhi = smem[cb + 2];
            v8i b;
            b[0] = blo.x; b[1] = blo.y; b[2] = blo.z; b[3] = blo.w;
            b[4] = bhi.x; b[5] = bhi.y; b[6] = bhi.z; b[7] = bhi.w;
            acc[s] = __builtin_amdgcn_wmma_i32_16x16x64_iu8(true, a, true, b, acc[s], false, false);
        }
        __syncthreads();
    }

    if (!active) return;
    int mofs = (lane & 16) ? 8 : 0;
    if (MODE == 0) {
        signed char* op = outB + (size_t)(tileM + mofs) * 128 + tileNbase + colin;
#pragma unroll
        for (int r = 0; r < 8; r++) {
#pragma unroll
            for (int s = 0; s < NT; s++) {
                int c = tileNbase + s * 16 + colin;
                float t = sc[c] * (float)acc[s][r] + sh[c];
                op[s * 16] = sgn8(t);
            }
            op += 128;
        }
    } else {
        int HW = H * W;
        int row0 = tileM + mofs;
        int nn = row0 / HW, rr = row0 % HW, yy = rr / W, xx = rr % W;
#pragma unroll
        for (int r = 0; r < 8; r++) {
            size_t base = (((size_t)nn * Cstride + c0 + tileNbase + colin) * H + yy) * W + xx;
#pragma unroll
            for (int s = 0; s < NT; s++)
                outF[base + (size_t)(s * 16) * HW] = (float)acc[s][r];
            xx++;
            if (xx == W) { xx = 0; yy++; if (yy == H) { yy = 0; nn++; } }
        }
    }
}

__global__ void bgemm3x3_lds_k(const signed char* __restrict__ A2, const signed char* __restrict__ Wb2,
                               int Mtiles, float* __restrict__ outF, int Cstride, int c0,
                               int H, int W) {
    __shared__ int4 smem[2304];
    for (int u = threadIdx.x; u < 2304; u += 256) smem[u] = ((const int4*)Wb2)[u];
    __syncthreads();

    int wave = threadIdx.x >> 5;
    int lane = threadIdx.x & 31;
    int tM = blockIdx.x * 8 + wave;
    bool active = (tM < Mtiles);
    int tileM = (active ? tM : (Mtiles - 1)) * 16;

    int HW = H * W;
    int rowA = tileM + (lane & 15);
    int n = rowA / HW, rm = rowA % HW, y = rm / W, x = rm % W;
    int aoff = (lane & 16) ? 16 : 0;
    int hl = (lane >> 4) & 1;
    int colin = lane & 15;
    int cb0 = colin * 72 + hl;
    int cb1 = (16 + colin) * 72 + hl;

    v8i acc[2];
    acc[0] = (v8i){0, 0, 0, 0, 0, 0, 0, 0};
    acc[1] = (v8i){0, 0, 0, 0, 0, 0, 0, 0};

#pragma unroll
    for (int tap = 0; tap < 9; tap++) {
        int yy = y + (tap / 3) - 1;
        int xx = x + (tap % 3) - 1;
        bool ok = ((unsigned)yy < (unsigned)H) && ((unsigned)xx < (unsigned)W);
        const signed char* arow = A2 + ((size_t)n * HW + yy * W + xx) * 128 + aoff;
#pragma unroll
        for (int half = 0; half < 2; half++) {
            int kq = tap * 8 + half * 4;
            v8i a = {0, 0, 0, 0, 0, 0, 0, 0};
            if (ok) {
                const signed char* ar = arow + half * 64;
                const int* ap;
                ap = (const int*)(ar);      a[0] = ap[0]; a[1] = ap[1];
                ap = (const int*)(ar + 8);  a[2] = ap[0]; a[3] = ap[1];
                ap = (const int*)(ar + 32); a[4] = ap[0]; a[5] = ap[1];
                ap = (const int*)(ar + 40); a[6] = ap[0]; a[7] = ap[1];
            }
            {
                int4 blo = smem[cb0 + kq], bhi = smem[cb0 + kq + 2];
                v8i b;
                b[0] = blo.x; b[1] = blo.y; b[2] = blo.z; b[3] = blo.w;
                b[4] = bhi.x; b[5] = bhi.y; b[6] = bhi.z; b[7] = bhi.w;
                acc[0] = __builtin_amdgcn_wmma_i32_16x16x64_iu8(true, a, true, b, acc[0], false, false);
            }
            {
                int4 blo = smem[cb1 + kq], bhi = smem[cb1 + kq + 2];
                v8i b;
                b[0] = blo.x; b[1] = blo.y; b[2] = blo.z; b[3] = blo.w;
                b[4] = bhi.x; b[5] = bhi.y; b[6] = bhi.z; b[7] = bhi.w;
                acc[1] = __builtin_amdgcn_wmma_i32_16x16x64_iu8(true, a, true, b, acc[1], false, false);
            }
        }
    }

    if (!active) return;
    int mofs = (lane & 16) ? 8 : 0;
    int row0 = tileM + mofs;
    int nn = row0 / HW, rr = row0 % HW, yy = rr / W, xx = rr % W;
#pragma unroll
    for (int r = 0; r < 8; r++) {
        size_t base = (((size_t)nn * Cstride + c0 + colin) * H + yy) * W + xx;
        outF[base] = (float)acc[0][r];
        outF[base + (size_t)16 * HW] = (float)acc[1][r];
        xx++;
        if (xx == W) { xx = 0; yy++; if (yy == H) { yy = 0; nn++; } }
    }
}

__global__ void bgemm_fc_k(const signed char* __restrict__ A, const signed char* __restrict__ Bw,
                           int Kpad, float* __restrict__ outF, const float* __restrict__ bias,
                           int Nreal, int ldN) {
    int lane = threadIdx.x & 31;
    int tileN = blockIdx.y * 16;
    const signed char* arow = A + (size_t)(lane & 15) * Kpad + ((lane & 16) ? 16 : 0);
    const signed char* brow = Bw + (size_t)(tileN + (lane & 15)) * Kpad + ((lane & 16) ? 16 : 0);
    v8i acc = {0, 0, 0, 0, 0, 0, 0, 0};
    for (int k0 = 0; k0 < Kpad; k0 += 64) {
        v8i a, b;
        const int* ap;
        ap = (const int*)(arow + k0);      a[0] = ap[0]; a[1] = ap[1];
        ap = (const int*)(arow + k0 + 8);  a[2] = ap[0]; a[3] = ap[1];
        ap = (const int*)(arow + k0 + 32); a[4] = ap[0]; a[5] = ap[1];
        ap = (const int*)(arow + k0 + 40); a[6] = ap[0]; a[7] = ap[1];
        const int* bp;
        bp = (const int*)(brow + k0);      b[0] = bp[0]; b[1] = bp[1]; b[2] = bp[2]; b[3] = bp[3];
        bp = (const int*)(brow + k0 + 32); b[4] = bp[0]; b[5] = bp[1]; b[6] = bp[2]; b[7] = bp[3];
        acc = __builtin_amdgcn_wmma_i32_16x16x64_iu8(true, a, true, b, acc, false, false);
    }
    int colin = lane & 15;
    int mofs = (lane & 16) ? 8 : 0;
#pragma unroll
    for (int r = 0; r < 8; r++) {
        int row = r + mofs;
        int c = tileN + colin;
        if (c < Nreal) outF[(size_t)row * ldN + c] = (float)acc[r] + bias[c];
    }
}

__global__ void avgpool2_k(const float* __restrict__ in, float* __restrict__ out,
                           int Cout, int H, int W, int H2, int W2, int Cnext) {
    int i = blockIdx.x * blockDim.x + threadIdx.x;
    size_t total = (size_t)16 * Cout * H2 * W2;
    if ((size_t)i >= total) return;
    int x2 = i % W2; int t = i / W2;
    int y2 = t % H2; t /= H2;
    int c = t % Cout; int nn = t / Cout;
    const float* p = in + (((size_t)nn * Cout + c) * H + 2 * y2) * W + 2 * x2;
    float s = p[0] + p[1] + p[W] + p[W + 1];
    out[(((size_t)nn * Cnext + c) * H2 + y2) * W2 + x2] = 0.25f * s;
}

__global__ void pool_bin_k(const float* __restrict__ h, const float* __restrict__ sc,
                           const float* __restrict__ sh, signed char* __restrict__ out) {
    int i = blockIdx.x * blockDim.x + threadIdx.x;
    if (i >= 16 * 1024) return;
    int c = i & 1023, n = i >> 10;
    const float* p = h + ((size_t)n * 1024 + c) * 49;
    float s = 0.f;
    float a = sc[c], bsh = sh[c];
    for (int q = 0; q < 49; q++) {
        float t = a * p[q] + bsh;
        s += fminf(1.f, fmaxf(-1.f, t));
    }
    out[i] = sgn8(s * (1.f / 49.f));
}

__global__ void softmax_k(const float* __restrict__ logits, float* __restrict__ probas) {
    __shared__ float red[256];
    int n = blockIdx.x, tid = threadIdx.x;
    const float* row = logits + (size_t)n * 1000;
    float mx = -3.4e38f;
    for (int j = tid; j < 1000; j += 256) mx = fmaxf(mx, row[j]);
    red[tid] = mx; __syncthreads();
    for (int s = 128; s > 0; s >>= 1) { if (tid < s) red[tid] = fmaxf(red[tid], red[tid + s]); __syncthreads(); }
    float m = red[0]; __syncthreads();
    float acc = 0.f;
    for (int j = tid; j < 1000; j += 256) acc += __expf(row[j] - m);
    red[tid] = acc; __syncthreads();
    for (int s = 128; s > 0; s >>= 1) { if (tid < s) red[tid] += red[tid + s]; __syncthreads(); }
    float inv = 1.f / red[0];
    for (int j = tid; j < 1000; j += 256) probas[(size_t)n * 1000 + j] = __expf(row[j] - m) * inv;
}

extern "C" void kernel_launch(void* const* d_in, const int* in_sizes, int n_in,
                              void* d_out, int out_size, void* d_ws, size_t ws_size,
                              hipStream_t stream) {
    (void)in_sizes; (void)n_in; (void)out_size; (void)ws_size;
    int p = 0;
    auto F = [&]() { return (const float*)d_in[p++]; };

    const float* x  = F();
    const float* w0 = F();
    const float* n0[4]; for (int t = 0; t < 4; t++) n0[t] = F();

    struct LayerP { const float* n1[4]; const float* w1; const float* n2[4]; const float* w2; };
    static const int nlayers[4] = {6, 12, 24, 16};
    LayerP layers[58]; int li = 0;
    for (int bi = 0; bi < 4; bi++)
        for (int j = 0; j < nlayers[bi]; j++) {
            LayerP& L = layers[li++];
            for (int t = 0; t < 4; t++) L.n1[t] = F();
            L.w1 = F();
            for (int t = 0; t < 4; t++) L.n2[t] = F();
            L.w2 = F();
        }
    struct TransP { const float* n[4]; const float* w; } trans[3];
    for (int i = 0; i < 3; i++) { for (int t = 0; t < 4; t++) trans[i].n[t] = F(); trans[i].w = F(); }
    const float* n5[4]; for (int t = 0; t < 4; t++) n5[t] = F();
    const float* fcw = F();
    const float* fcb = F();

    char* ws = (char*)d_ws; size_t off = 0;
    auto carve = [&](size_t bytes) { void* r = ws + off; off = (off + bytes + 255) & ~(size_t)255; return r; };
    float* hA   = (float*)carve((size_t)16 * 256 * 57 * 57 * 4);
    float* hB   = (float*)carve((size_t)16 * 256 * 57 * 57 * 4);
    float* tbuf = (float*)carve((size_t)16 * 128 * 57 * 57 * 4);
    signed char* abin = (signed char*)carve((size_t)51984 * 256);
    signed char* ab2  = (signed char*)carve((size_t)51984 * 128);
    signed char* wb   = (signed char*)carve((size_t)1008 * 1152);
    float* sc1 = (float*)carve(1024 * 4); float* sh1 = (float*)carve(1024 * 4);
    float* sc2 = (float*)carve(1024 * 4); float* sh2 = (float*)carve(1024 * 4);
    signed char* pooled = (signed char*)carve(16 * 1024);

    const int T = 256;
    auto nb = [&](size_t n) { return (unsigned)((n + T - 1) / T); };

    bn_fold_k<<<nb(64), T, 0, stream>>>(n0[0], n0[1], n0[2], n0[3], sc1, sh1, 64);
    conv0_k<<<nb((size_t)16 * 64 * 114 * 114), T, 0, stream>>>(x, w0, sc1, sh1, hB);
    maxpool_k<<<nb((size_t)16 * 64 * 57 * 57), T, 0, stream>>>(hB, hA, 256);

    float* hcur = hA; float* hoth = hB;
    int H = 57, W = 57, cin = 64;
    static const int blkC[4] = {256, 512, 1024, 1024};
    li = 0;

    for (int bi = 0; bi < 4; bi++) {
        int Cblk = blkC[bi];
        int M = 16 * H * W, Mtiles = M / 16;
        unsigned gx = (unsigned)((Mtiles + 7) / 8);
        for (int j = 0; j < nlayers[bi]; j++) {
            LayerP& L = layers[li++];
            int Kpad = (cin + 63) & ~63;
            bn_fold_k<<<nb(cin), T, 0, stream>>>(L.n1[0], L.n1[1], L.n1[2], L.n1[3], sc1, sh1, cin);
            bn_fold_k<<<nb(128), T, 0, stream>>>(L.n2[0], L.n2[1], L.n2[2], L.n2[3], sc2, sh2, 128);
            binw1_k<<<nb((size_t)128 * Kpad), T, 0, stream>>>(L.w1, wb, 128, 128, cin, Kpad);
            binarize_h_k<<<nb((size_t)M * Kpad), T, 0, stream>>>(hcur, sc1, sh1, abin, cin, Kpad, Cblk, H, W);
            bgemm_lds_k<0, 8><<<dim3(gx, 1), T, 0, stream>>>(abin, wb, Mtiles, Kpad, sc2, sh2, ab2,
                                                             nullptr, 0, 0, H, W);
            binw2_k<<<nb((size_t)32 * 1152), T, 0, stream>>>(L.w2, wb);
            bgemm3x3_lds_k<<<dim3(gx, 1), T, 0, stream>>>(ab2, wb, Mtiles, hcur, Cblk, cin, H, W);
            cin += 32;
        }
        if (bi < 3) {
            TransP& Tr = trans[bi];
            int Kpad = (cin + 63) & ~63;
            int Cout = cin / 2;
            bn_fold_k<<<nb(cin), T, 0, stream>>>(Tr.n[0], Tr.n[1], Tr.n[2], Tr.n[3], sc1, sh1, cin);
            binw1_k<<<nb((size_t)Cout * Kpad), T, 0, stream>>>(Tr.w, wb, Cout, Cout, cin, Kpad);
            binarize_h_k<<<nb((size_t)M * Kpad), T, 0, stream>>>(hcur, sc1, sh1, abin, cin, Kpad, Cblk, H, W);
            bgemm_lds_k<1, 8><<<dim3(gx, (unsigned)(Cout / 128)), T, 0, stream>>>(
                abin, wb, Mtiles, Kpad, nullptr, nullptr, nullptr, tbuf, Cout, 0, H, W);
            int H2 = H / 2, W2 = W / 2;
            avgpool2_k<<<nb((size_t)16 * Cout * H2 * W2), T, 0, stream>>>(tbuf, hoth, Cout, H, W, H2, W2, blkC[bi + 1]);
            float* tmp = hcur; hcur = hoth; hoth = tmp;
            H = H2; W = W2; cin = Cout;
        }
    }

    bn_fold_k<<<nb(1024), T, 0, stream>>>(n5[0], n5[1], n5[2], n5[3], sc1, sh1, 1024);
    pool_bin_k<<<nb((size_t)16 * 1024), T, 0, stream>>>(hcur, sc1, sh1, pooled);
    binw1_k<<<nb((size_t)1008 * 1024), T, 0, stream>>>(fcw, wb, 1008, 1000, 1024, 1024);
    float* logits = (float*)d_out;
    float* probas = logits + 16 * 1000;
    bgemm_fc_k<<<dim3(1, 63), 32, 0, stream>>>(pooled, wb, 1024, logits, fcb, 1000, 1000);
    softmax_k<<<16, T, 0, stream>>>(logits, probas);
}
